// Sequence_43147241455818
// MI455X (gfx1250) — matrix operand order repaired and checked by reading, not compiled, not run on hardware
//
#include <hip/hip_runtime.h>

typedef __attribute__((ext_vector_type(16))) _Float16 v16h;
typedef __attribute__((ext_vector_type(8)))  _Float16 v8h;
typedef __attribute__((ext_vector_type(4)))  _Float16 v4h;
typedef __attribute__((ext_vector_type(8)))  float    v8f;
typedef __attribute__((ext_vector_type(4)))  float    v4f;

#define HID   51
#define H4    204
#define TSTEP 1000
#define BATCH 1024
#define MB    16      // batch rows per block (one wave32)
#define NTIL  13      // 13 * 16 = 208 >= 204 gate columns
#define WS    72      // padded K stride in halves (bank-conflict-free b128 frags)
#define GS    20      // gatesT row stride in floats (16B-aligned b128 stores)
#define NACT  26      // ceil(16*51 / 32) activation units per lane

// ---------------- LDS layout (bytes, all offsets 16B aligned) ----------------
constexpr unsigned SZ_W    = 208u * WS * 2u;              // 29952 per weight mat
constexpr unsigned OFF_W1  = 0u;                          // Whh1 [n][k]
constexpr unsigned OFF_W2I = OFF_W1  + SZ_W;              // Wih2 [n][k]
constexpr unsigned OFF_W2H = OFF_W2I + SZ_W;              // Whh2 [n][k]
constexpr unsigned OFF_H1  = OFF_W2H + SZ_W;              // h1 f16 [16][WS]
constexpr unsigned OFF_H2  = OFF_H1  + MB * WS * 2u;      // h2 f16 [16][WS]
constexpr unsigned OFF_G   = OFF_H2  + MB * WS * 2u;      // gatesT f32 [208][GS]
constexpr unsigned OFF_B1  = OFF_G   + 208u * GS * 4u;    // bias1 f32 [204]
constexpr unsigned OFF_B2  = OFF_B1  + 832u;              // bias2 f32 [204]
constexpr unsigned OFF_WI  = OFF_B2  + 832u;              // W_ih1 col f32 [204]
constexpr unsigned OFF_WL  = OFF_WI  + 832u;              // W_lin f32 [52]
constexpr unsigned OFF_X   = OFF_WL  + 224u;              // x slice f32 [16][1000]
constexpr unsigned LDS_BYTES = OFF_X + MB * TSTEP * 4u;   // ~176 KB (< 320 KB/WGP)

__device__ __forceinline__ float fsig(float x) {
    return __builtin_amdgcn_rcpf(1.0f + __expf(-x));
}
__device__ __forceinline__ float ftanh(float x) {
    float e = __expf(2.0f * x);
    return 1.0f - 2.0f * __builtin_amdgcn_rcpf(e + 1.0f);
}

// A-matrix 16x32 f16 fragment: lanes 0-15 hold K[0..7],K[16..23]; lanes 16-31
// hold K[8..15],K[24..31] (dA = 0 or 8). hbuf is row-major [16][WS] halves.
__device__ __forceinline__ v16h fragA(const _Float16* hbuf, int lm, int k0, int dA) {
    const _Float16* p = hbuf + lm * WS + k0 + dA;
    v8h lo = *(const v8h*)(p);
    v8h hi = *(const v8h*)(p + 16);
    return __builtin_shufflevector(lo, hi, 0,1,2,3,4,5,6,7,8,9,10,11,12,13,14,15);
}
// B-matrix 32x16 f16 fragment: lane n holds column n, K[0..7],K[16..23]; lane n+16
// holds K[8..15],K[24..31] (dB = 0 or 8). wbuf is [col][WS] halves (K contiguous per column).
__device__ __forceinline__ v16h fragB(const _Float16* wbuf, int col, int k0, int dB) {
    const _Float16* p = wbuf + col * WS + k0 + dB;
    v8h lo = *(const v8h*)(p);
    v8h hi = *(const v8h*)(p + 16);
    return __builtin_shufflevector(lo, hi, 0,1,2,3,4,5,6,7,8,9,10,11,12,13,14,15);
}
__device__ __forceinline__ v8f wmma_f16(v16h a, v16h b, v8f c) {
    return __builtin_amdgcn_wmma_f32_16x16x32_f16(false, a, false, b, (short)0, c,
                                                  false, false);
}
__device__ __forceinline__ void storeD(float* gT, int col, int dA, v8f acc) {
    float* gp = gT + col * GS + dA;
    *(v4f*)(gp)     = __builtin_shufflevector(acc, acc, 0, 1, 2, 3);
    *(v4f*)(gp + 4) = __builtin_shufflevector(acc, acc, 4, 5, 6, 7);
}

__global__ __launch_bounds__(32, 1)
void lstm2_persistent(const float* __restrict__ x,
                      const float* __restrict__ Wih1, const float* __restrict__ Whh1,
                      const float* __restrict__ bih1, const float* __restrict__ bhh1,
                      const float* __restrict__ Wih2, const float* __restrict__ Whh2,
                      const float* __restrict__ bih2, const float* __restrict__ bhh2,
                      const float* __restrict__ Wlin, const float* __restrict__ blin,
                      float* __restrict__ out) {
    extern __shared__ char smem[];
    _Float16* w1  = (_Float16*)(smem + OFF_W1);
    _Float16* w2i = (_Float16*)(smem + OFF_W2I);
    _Float16* w2h = (_Float16*)(smem + OFF_W2H);
    _Float16* h1p = (_Float16*)(smem + OFF_H1);
    _Float16* h2p = (_Float16*)(smem + OFF_H2);
    float*    gT  = (float*)(smem + OFF_G);
    float*    b1p = (float*)(smem + OFF_B1);
    float*    b2p = (float*)(smem + OFF_B2);
    float*    wi1 = (float*)(smem + OFF_WI);
    float*    wlp = (float*)(smem + OFF_WL);
    float*    xs  = (float*)(smem + OFF_X);

    const int lane = threadIdx.x;      // 0..31, single wave per block
    const int lm   = lane & 15;
    const int dA   = (lane & 16) ? 8  : 0;   // A fragment K interleave
    const int dB   = (lane & 16) ? 8  : 0;   // B fragment K interleave

    // ---- one-time init: weights into LDS in fragment layout [col][K] ----
    for (int i = lane; i < 208 * WS; i += 32) {
        int n = i / WS, k = i - n * WS;
        bool v = (n < H4) && (k < HID);
        w1[i]  = v ? (_Float16)Whh1[n * HID + k] : (_Float16)0.0f;
        w2i[i] = v ? (_Float16)Wih2[n * HID + k] : (_Float16)0.0f;
        w2h[i] = v ? (_Float16)Whh2[n * HID + k] : (_Float16)0.0f;
    }
    for (int i = lane; i < MB * WS; i += 32) { h1p[i] = (_Float16)0.0f; h2p[i] = (_Float16)0.0f; }
    for (int i = lane; i < H4; i += 32) {
        b1p[i] = bih1[i] + bhh1[i];
        b2p[i] = bih2[i] + bhh2[i];
        wi1[i] = Wih1[i];
    }
    for (int i = lane; i < 52; i += 32) wlp[i] = (i < HID) ? Wlin[i] : 0.0f;
    // whole input slice for this block's 16 batch rows (contiguous in global)
    {
        const v4f* src = (const v4f*)(x + (size_t)blockIdx.x * MB * TSTEP);
        v4f* dst = (v4f*)xs;
        for (int i = lane; i < MB * TSTEP / 4; i += 32) dst[i] = src[i];
    }
    const float bl = blin[0];
    const int   obase = blockIdx.x * MB;

    float c1[NACT], c2[NACT];
#pragma unroll
    for (int i = 0; i < NACT; ++i) { c1[i] = 0.0f; c2[i] = 0.0f; }

    const v8f vzero = {0.f,0.f,0.f,0.f,0.f,0.f,0.f,0.f};

    for (int t = 0; t < TSTEP; ++t) {
        // ================= layer 1: gates = h1_old @ Whh1^T =================
        // Stream of 26 WMMAs (13 tiles x 2 K-chunks), depth-2 B prefetch so the
        // s_wait before each WMMA is partial instead of a full dscnt drain.
        {
            v16h a0 = fragA(h1p, lm, 0, dA);
            v16h a1 = fragA(h1p, lm, 32, dA);
            v16h f0 = fragB(w1, lm, 0, dB);        // s = 0: tile 0, k0 = 0
            v16h f1 = fragB(w1, lm, 32, dB);       // s = 1: tile 0, k0 = 32
            v8f acc = vzero;
#pragma unroll
            for (int s = 0; s < 2 * NTIL; ++s) {
                const int nt = s >> 1, ph = s & 1;
                v16h fn = f0;
                if (s + 2 < 2 * NTIL)              // prefetch fragment for s+2
                    fn = fragB(w1, (nt + 1) * 16 + lm, ph * 32, dB);
                if (ph == 0) {
                    acc = wmma_f16(a0, f0, vzero);
                } else {
                    acc = wmma_f16(a1, f0, acc);
                    storeD(gT, nt * 16 + lm, dA, acc);
                }
                f0 = f1; f1 = fn;
            }
        }
        // ---- layer-1 activations (adds x*W_ih1 rank-1 term + biases) ----
#pragma unroll
        for (int it = 0; it < NACT; ++it) {
            int u = lane + it * 32;
            if (u < MB * HID) {
                int j = u >> 4, m = u & 15;
                float xi = xs[m * TSTEP + t];
                float gi = gT[(j          ) * GS + m] + xi * wi1[j          ] + b1p[j          ];
                float gf = gT[(j +     HID) * GS + m] + xi * wi1[j +     HID] + b1p[j +     HID];
                float gg = gT[(j + 2 * HID) * GS + m] + xi * wi1[j + 2 * HID] + b1p[j + 2 * HID];
                float go = gT[(j + 3 * HID) * GS + m] + xi * wi1[j + 3 * HID] + b1p[j + 3 * HID];
                float cn = fsig(gf) * c1[it] + fsig(gi) * ftanh(gg);
                c1[it] = cn;
                h1p[m * WS + j] = (_Float16)(fsig(go) * ftanh(cn));
            }
        }
        // ========= layer 2: gates = h1_new @ Wih2^T + h2_old @ Whh2^T =========
        // Stream of 52 WMMAs (13 tiles x {w2i,w2h} x 2 K-chunks), depth-2 prefetch.
        {
            v16h p0 = fragA(h1p, lm, 0, dA);
            v16h p1 = fragA(h1p, lm, 32, dA);
            v16h q0 = fragA(h2p, lm, 0, dA);
            v16h q1 = fragA(h2p, lm, 32, dA);
            v16h f0 = fragB(w2i, lm, 0, dB);       // s = 0
            v16h f1 = fragB(w2i, lm, 32, dB);      // s = 1
            v8f acc = vzero;
#pragma unroll
            for (int s = 0; s < 4 * NTIL; ++s) {
                const int nt = s >> 2, ph = s & 3;
                v16h fn = f0;
                if (s + 2 < 4 * NTIL) {            // prefetch fragment for s+2
                    const int s2 = s + 2, nt2 = s2 >> 2, ph2 = s2 & 3;
                    const _Float16* wb = (ph2 < 2) ? w2i : w2h;
                    fn = fragB(wb, nt2 * 16 + lm, (ph2 & 1) * 32, dB);
                }
                if      (ph == 0) acc = wmma_f16(p0, f0, vzero);
                else if (ph == 1) acc = wmma_f16(p1, f0, acc);
                else if (ph == 2) acc = wmma_f16(q0, f0, acc);
                else {
                    acc = wmma_f16(q1, f0, acc);
                    storeD(gT, nt * 16 + lm, dA, acc);
                }
                f0 = f1; f1 = fn;
            }
        }
        // ---- layer-2 activations ----
#pragma unroll
        for (int it = 0; it < NACT; ++it) {
            int u = lane + it * 32;
            if (u < MB * HID) {
                int j = u >> 4, m = u & 15;
                float gi = gT[(j          ) * GS + m] + b2p[j          ];
                float gf = gT[(j +     HID) * GS + m] + b2p[j +     HID];
                float gg = gT[(j + 2 * HID) * GS + m] + b2p[j + 2 * HID];
                float go = gT[(j + 3 * HID) * GS + m] + b2p[j + 3 * HID];
                float cn = fsig(gf) * c2[it] + fsig(gi) * ftanh(gg);
                c2[it] = cn;
                h2p[m * WS + j] = (_Float16)(fsig(go) * ftanh(cn));
            }
        }
        // ---- out[:, t] = h2 @ W_lin^T + b_lin (lanes 0..15, one row each) ----
        if (lane < 16) {
            float acc = bl;
#pragma unroll
            for (int j0 = 0; j0 < 52; j0 += 4) {
                v4h hv = *(const v4h*)(h2p + lane * WS + j0);
                acc += (float)hv[0] * wlp[j0]     + (float)hv[1] * wlp[j0 + 1] +
                       (float)hv[2] * wlp[j0 + 2] + (float)hv[3] * wlp[j0 + 3];
            }
            out[(size_t)(obase + lane) * TSTEP + t] = acc;
        }
    }
}

extern "C" void kernel_launch(void* const* d_in, const int* in_sizes, int n_in,
                              void* d_out, int out_size, void* d_ws, size_t ws_size,
                              hipStream_t stream) {
    const float* x    = (const float*)d_in[0];
    const float* Wih1 = (const float*)d_in[1];
    const float* Whh1 = (const float*)d_in[2];
    const float* bih1 = (const float*)d_in[3];
    const float* bhh1 = (const float*)d_in[4];
    const float* Wih2 = (const float*)d_in[5];
    const float* Whh2 = (const float*)d_in[6];
    const float* bih2 = (const float*)d_in[7];
    const float* bhh2 = (const float*)d_in[8];
    const float* Wlin = (const float*)d_in[9];
    const float* blin = (const float*)d_in[10];
    float* out = (float*)d_out;

    (void)hipFuncSetAttribute((const void*)lstm2_persistent,
                              hipFuncAttributeMaxDynamicSharedMemorySize,
                              (int)LDS_BYTES);
    dim3 grid(BATCH / MB);   // 64 independent single-wave recurrence streams
    dim3 block(32);          // one wave32 per block -> no barriers needed
    lstm2_persistent<<<grid, block, LDS_BYTES, stream>>>(
        x, Wih1, Whh1, bih1, bhh1, Wih2, Whh2, bih2, bhh2, Wlin, blin, out);
}
